// MOSSAttention_41970420418039
// MI455X (gfx1250) — compile-verified, matrix operand order checked by reading, not run on hardware
//
#include <hip/hip_runtime.h>
#include <hip/hip_bf16.h>

// ---------- types ----------
typedef __attribute__((ext_vector_type(16))) __bf16 bf16x16;
typedef __attribute__((ext_vector_type(2)))  __bf16 bf16x2;
typedef __attribute__((ext_vector_type(8)))  float  f32x8;

union Frag { uint4 u[2]; bf16x16 v; };

#if __has_builtin(__builtin_amdgcn_cvt_pk_bf16_f32)
#define HAVE_CVT_PK_BF16 1
#endif

// pack two f32 -> two bf16 (RTNE)
__device__ __forceinline__ unsigned pk2(float a, float b) {
#ifdef HAVE_CVT_PK_BF16
    union { bf16x2 v; unsigned u; } c;
    c.v = __builtin_amdgcn_cvt_pk_bf16_f32(a, b);
    return c.u;
#else
    union { float f; unsigned u; } ca, cb; ca.f = a; cb.f = b;
    unsigned ra = (ca.u + 0x7FFFu + ((ca.u >> 16) & 1u)) >> 16;
    unsigned rb = (cb.u + 0x7FFFu + ((cb.u >> 16) & 1u)) >> 16;
    return (ra & 0xFFFFu) | (rb << 16);
#endif
}
__device__ __forceinline__ unsigned short f2bf(float x) {
#ifdef HAVE_CVT_PK_BF16
    return (unsigned short)(pk2(x, x) & 0xFFFFu);
#else
    union { float f; unsigned u; } c; c.f = x;
    return (unsigned short)((c.u + 0x7FFFu + ((c.u >> 16) & 1u)) >> 16);
#endif
}

// Load a 16x32 bf16 A/B fragment from a row-major (K-contiguous) row.
// Per ISA 7.12.2: lanes 0-15 hold K = [0..7, 16..23], lanes 16-31 hold K = [8..15, 24..31].
__device__ __forceinline__ bf16x16 ld_frag(const unsigned short* row, int k0, int hlf) {
    Frag f;
    f.u[0] = *(const uint4*)(row + k0 + 8 * hlf);
    f.u[1] = *(const uint4*)(row + k0 + 16 + 8 * hlf);
    return f.v;
}

__device__ __forceinline__ f32x8 wmma_bf16(bf16x16 a, bf16x16 b, f32x8 c) {
    return __builtin_amdgcn_wmma_f32_16x16x32_bf16(false, a, false, b, (short)0, c, false, false);
}

#define SEQ 2048
#define HID 4096
#define NHEAD 32
#define HDIM 128
#define LDSK 72   // 64 + 8 pad: rows stay 16B aligned (144B) and banks spread

// =====================================================================
// Kernel 1: fused QKV GEMM (bf16 WMMA, f32 accum) + bias + RoPE
// grid = (M/128=32, 3*NH=96), block = 256 (8 waves; wave w -> rows w*16.., cols 0..127)
// Single-buffered LDS (36KB) -> several workgroups co-resident per WGP hide latency.
// =====================================================================
__global__ __launch_bounds__(256) void qkv_rope_kernel(
    const float* __restrict__ hs, const float* __restrict__ wqkv,
    const float* __restrict__ bqkv, const int* __restrict__ pos_ids,
    unsigned short* __restrict__ Qd, unsigned short* __restrict__ Kd,
    unsigned short* __restrict__ Vd)
{
    __shared__ unsigned short As[128 * LDSK];
    __shared__ unsigned short Bs[128 * LDSK];
    const int tid  = threadIdx.x;
    const int wave = tid >> 5, lane = tid & 31;
    const int hlf  = lane >> 4, lcol = lane & 15;
    const int mbase = blockIdx.x * 128;
    const int gy    = blockIdx.y;            // section*32 + head
    const int nbase = gy * 128;

    f32x8 acc[8];
    #pragma unroll
    for (int t = 0; t < 8; ++t)
        #pragma unroll
        for (int j = 0; j < 8; ++j) acc[t][j] = 0.f;

    for (int kt = 0; kt < HID; kt += 64) {
        #pragma unroll
        for (int i = 0; i < 8; ++i) {
            int idx = tid + i * 256;          // 0..2047
            int r = idx >> 4, c4 = idx & 15;
            float4 a = *(const float4*)(hs + (size_t)(mbase + r) * HID + kt + c4 * 4);
            uint2 pa; pa.x = pk2(a.x, a.y); pa.y = pk2(a.z, a.w);
            *(uint2*)(As + r * LDSK + c4 * 4) = pa;
            float4 b = *(const float4*)(wqkv + (size_t)(nbase + r) * HID + kt + c4 * 4);
            uint2 pb; pb.x = pk2(b.x, b.y); pb.y = pk2(b.z, b.w);
            *(uint2*)(Bs + r * LDSK + c4 * 4) = pb;
        }
        __syncthreads();
        #pragma unroll
        for (int ks = 0; ks < 2; ++ks) {
            bf16x16 af = ld_frag(As + (wave * 16 + lcol) * LDSK, ks * 32, hlf);
            #pragma unroll
            for (int nt = 0; nt < 8; ++nt) {
                bf16x16 bfr = ld_frag(Bs + (nt * 16 + lcol) * LDSK, ks * 32, hlf);
                acc[nt] = wmma_bf16(af, bfr, acc[nt]);
            }
        }
        __syncthreads();
    }

    #pragma unroll
    for (int nt = 0; nt < 8; ++nt) {
        float bias = bqkv[nbase + nt * 16 + lcol];
        #pragma unroll
        for (int j = 0; j < 8; ++j) acc[nt][j] += bias;
    }

    const int section = gy >> 5, head = gy & 31;
    const int b = mbase / SEQ;                       // 128 | 2048 -> uniform
    unsigned short* dst = (section == 0) ? Qd : (section == 1) ? Kd : Vd;
    const size_t hbase = ((size_t)(b * NHEAD + head)) * SEQ * HDIM;
    const int srow0 = (mbase % SEQ) + wave * 16 + 8 * hlf;

    if (section < 2) {
        // NeoX RoPE: pair (i, i+64), angle = pos * theta^(-2i/128); pair lives in same lane
        float posf[8];
        #pragma unroll
        for (int j = 0; j < 8; ++j)
            posf[j] = (float)pos_ids[b * SEQ + srow0 + j];
        #pragma unroll
        for (int t = 0; t < 4; ++t) {
            int i = t * 16 + lcol;                   // 0..63
            float invf = __expf((float)i * (-2.0f / 128.0f) * 9.210340371976184f);
            #pragma unroll
            for (int j = 0; j < 8; ++j) {
                float sn, cs; __sincosf(posf[j] * invf, &sn, &cs);
                float x1 = acc[t][j], x2 = acc[t + 4][j];
                size_t o = hbase + (size_t)(srow0 + j) * HDIM;
                dst[o + i]      = f2bf(x1 * cs - x2 * sn);
                dst[o + i + 64] = f2bf(x2 * cs + x1 * sn);
            }
        }
    } else {
        #pragma unroll
        for (int t = 0; t < 8; ++t) {
            int i = t * 16 + lcol;
            #pragma unroll
            for (int j = 0; j < 8; ++j)
                dst[hbase + (size_t)(srow0 + j) * HDIM + i] = f2bf(acc[t][j]);
        }
    }
}

// =====================================================================
// Kernel 2: causal flash attention, bf16 WMMA / f32 online softmax
// grid = (S/64=32, B*NH=64), block = 128 (4 waves x 16 query rows)
// =====================================================================
__global__ __launch_bounds__(128) void attn_kernel(
    const unsigned short* __restrict__ Qd, const unsigned short* __restrict__ Kd,
    const unsigned short* __restrict__ Vd, unsigned short* __restrict__ ctx)
{
    __shared__ unsigned short Pl[64 * LDSK];    // P tile (C-layout -> A-layout round trip)
    __shared__ unsigned short Vt[128 * LDSK];   // V transposed: [hd][key]
    const int tid  = threadIdx.x;
    const int wave = tid >> 5, lane = tid & 31;
    const int hlf  = lane >> 4, lcol = lane & 15;
    const int qb = blockIdx.x;
    const int bh = blockIdx.y;
    const size_t base = (size_t)bh * SEQ * HDIM;
    const float scale = 0.08838834764831845f;   // 1/sqrt(128)

    bf16x16 qf[4];
    {
        const unsigned short* qrow = Qd + base + (size_t)(qb * 64 + wave * 16 + lcol) * HDIM;
        #pragma unroll
        for (int ks = 0; ks < 4; ++ks) qf[ks] = ld_frag(qrow, ks * 32, hlf);
    }
    f32x8 o[8];
    #pragma unroll
    for (int t = 0; t < 8; ++t)
        #pragma unroll
        for (int j = 0; j < 8; ++j) o[t][j] = 0.f;
    float m[8], l[8];
    #pragma unroll
    for (int j = 0; j < 8; ++j) { m[j] = -1e30f; l[j] = 0.f; }

    const int rowb = qb * 64 + wave * 16 + 8 * hlf;   // + j = absolute query row

    for (int jb = 0; jb <= qb; ++jb) {
        // S = Q K^T  (K fragments straight from global: rows are K-contiguous)
        f32x8 sc[4];
        #pragma unroll
        for (int nt = 0; nt < 4; ++nt) {
            #pragma unroll
            for (int j = 0; j < 8; ++j) sc[nt][j] = 0.f;
            const unsigned short* krow = Kd + base + (size_t)(jb * 64 + nt * 16 + lcol) * HDIM;
            #pragma unroll
            for (int ks = 0; ks < 4; ++ks) {
                bf16x16 kf = ld_frag(krow, ks * 32, hlf);
                sc[nt] = wmma_bf16(qf[ks], kf, sc[nt]);
            }
        }
        // scale + causal mask
        #pragma unroll
        for (int nt = 0; nt < 4; ++nt) {
            int col = jb * 64 + nt * 16 + lcol;
            #pragma unroll
            for (int j = 0; j < 8; ++j) {
                float s = sc[nt][j] * scale;
                sc[nt][j] = (col <= rowb + j) ? s : -1e30f;
            }
        }
        // online softmax (row = across 16 lanes of a half -> shfl_xor width 16)
        float mnew[8];
        #pragma unroll
        for (int j = 0; j < 8; ++j) {
            float v = sc[0][j];
            #pragma unroll
            for (int nt = 1; nt < 4; ++nt) v = fmaxf(v, sc[nt][j]);
            #pragma unroll
            for (int off = 8; off >= 1; off >>= 1) v = fmaxf(v, __shfl_xor(v, off, 16));
            mnew[j] = fmaxf(m[j], v);
        }
        float corr[8];
        #pragma unroll
        for (int j = 0; j < 8; ++j) corr[j] = __expf(m[j] - mnew[j]);
        #pragma unroll
        for (int t = 0; t < 8; ++t)
            #pragma unroll
            for (int j = 0; j < 8; ++j) o[t][j] *= corr[j];
        #pragma unroll
        for (int j = 0; j < 8; ++j) {
            float sum = 0.f;
            #pragma unroll
            for (int nt = 0; nt < 4; ++nt) {
                float p = __expf(sc[nt][j] - mnew[j]);
                sc[nt][j] = p;
                sum += p;
            }
            #pragma unroll
            for (int off = 8; off >= 1; off >>= 1) sum += __shfl_xor(sum, off, 16);
            l[j] = l[j] * corr[j] + sum;
            m[j] = mnew[j];
        }
        __syncthreads();   // prior P/Vt consumers done before overwrite
        // P (C-layout in regs) -> LDS row-major so it reloads as A fragments
        #pragma unroll
        for (int nt = 0; nt < 4; ++nt)
            #pragma unroll
            for (int j = 0; j < 8; ++j)
                Pl[(wave * 16 + j + 8 * hlf) * LDSK + nt * 16 + lcol] = f2bf(sc[nt][j]);
        // stage V transposed: Vt[hd][key]
        #pragma unroll
        for (int i = 0; i < 8; ++i) {
            int idx = tid + i * 128;            // 0..1023
            int key = idx >> 4, ch = idx & 15;
            uint4 v4 = *(const uint4*)(Vd + base + (size_t)(jb * 64 + key) * HDIM + ch * 8);
            const unsigned short* ve = (const unsigned short*)&v4;
            #pragma unroll
            for (int e = 0; e < 8; ++e) Vt[(ch * 8 + e) * LDSK + key] = ve[e];
        }
        __syncthreads();
        // O += P V
        #pragma unroll
        for (int ks = 0; ks < 2; ++ks) {
            bf16x16 pf = ld_frag(Pl + (wave * 16 + lcol) * LDSK, ks * 32, hlf);
            #pragma unroll
            for (int ht = 0; ht < 8; ++ht) {
                bf16x16 vf = ld_frag(Vt + (ht * 16 + lcol) * LDSK, ks * 32, hlf);
                o[ht] = wmma_bf16(pf, vf, o[ht]);
            }
        }
    }
    // normalize and write ctx [B,S,H] as bf16 (input to output projection)
    const int b = bh >> 5, h = bh & 31;
    #pragma unroll
    for (int j = 0; j < 8; ++j) {
        float inv = 1.0f / l[j];
        int srow = qb * 64 + wave * 16 + j + 8 * hlf;
        size_t ob = ((size_t)(b * SEQ + srow)) * HID + h * HDIM;
        #pragma unroll
        for (int ht = 0; ht < 8; ++ht)
            ctx[ob + ht * 16 + lcol] = f2bf(o[ht][j] * inv);
    }
}

// =====================================================================
// Kernel 3: output projection  out = ctx @ w_out^T + b_out  (fp32 out)
// grid = (32, 32), block = 256, single-buffered LDS
// =====================================================================
__global__ __launch_bounds__(256) void out_proj_kernel(
    const unsigned short* __restrict__ ctx, const float* __restrict__ wout,
    const float* __restrict__ bout, float* __restrict__ out)
{
    __shared__ unsigned short As[128 * LDSK];
    __shared__ unsigned short Bs[128 * LDSK];
    const int tid  = threadIdx.x;
    const int wave = tid >> 5, lane = tid & 31;
    const int hlf  = lane >> 4, lcol = lane & 15;
    const int mbase = blockIdx.x * 128;
    const int nbase = blockIdx.y * 128;

    f32x8 acc[8];
    #pragma unroll
    for (int t = 0; t < 8; ++t)
        #pragma unroll
        for (int j = 0; j < 8; ++j) acc[t][j] = 0.f;

    for (int kt = 0; kt < HID; kt += 64) {
        #pragma unroll
        for (int i = 0; i < 4; ++i) {          // A: already bf16, 1024 uint4 chunks
            int idx = tid + i * 256;
            int r = idx >> 3, c = idx & 7;
            *(uint4*)(As + r * LDSK + c * 8) =
                *(const uint4*)(ctx + (size_t)(mbase + r) * HID + kt + c * 8);
        }
        #pragma unroll
        for (int i = 0; i < 8; ++i) {          // B: fp32 -> bf16
            int idx = tid + i * 256;
            int r = idx >> 4, c4 = idx & 15;
            float4 b = *(const float4*)(wout + (size_t)(nbase + r) * HID + kt + c4 * 4);
            uint2 pb; pb.x = pk2(b.x, b.y); pb.y = pk2(b.z, b.w);
            *(uint2*)(Bs + r * LDSK + c4 * 4) = pb;
        }
        __syncthreads();
        #pragma unroll
        for (int ks = 0; ks < 2; ++ks) {
            bf16x16 af = ld_frag(As + (wave * 16 + lcol) * LDSK, ks * 32, hlf);
            #pragma unroll
            for (int nt = 0; nt < 8; ++nt) {
                bf16x16 bfr = ld_frag(Bs + (nt * 16 + lcol) * LDSK, ks * 32, hlf);
                acc[nt] = wmma_bf16(af, bfr, acc[nt]);
            }
        }
        __syncthreads();
    }

    #pragma unroll
    for (int nt = 0; nt < 8; ++nt) {
        float bias = bout[nbase + nt * 16 + lcol];
        #pragma unroll
        for (int j = 0; j < 8; ++j) {
            int row = mbase + wave * 16 + j + 8 * hlf;
            out[(size_t)row * HID + nbase + nt * 16 + lcol] = acc[nt][j] + bias;
        }
    }
}

// =====================================================================
extern "C" void kernel_launch(void* const* d_in, const int* in_sizes, int n_in,
                              void* d_out, int out_size, void* d_ws, size_t ws_size,
                              hipStream_t stream) {
    const int*   pos  = (const int*)d_in[0];
    const float* hs   = (const float*)d_in[1];
    const float* wqkv = (const float*)d_in[2];
    const float* bqkv = (const float*)d_in[3];
    const float* wout = (const float*)d_in[4];
    const float* bout = (const float*)d_in[5];

    const size_t PER = (size_t)2 * NHEAD * SEQ * HDIM;   // 16,777,216 elems (bf16)
    unsigned short* Q = (unsigned short*)d_ws;
    unsigned short* K = Q + PER;
    unsigned short* V = K + PER;
    unsigned short* C = V + PER;                         // ctx, 128 MiB total ws

    qkv_rope_kernel<<<dim3(32, 96), 256, 0, stream>>>(hs, wqkv, bqkv, pos, Q, K, V);
    attn_kernel<<<dim3(32, 64), 128, 0, stream>>>(Q, K, V, C);
    out_proj_kernel<<<dim3(32, 32), 256, 0, stream>>>(C, wout, bout, (float*)d_out);
}
